// HGTModel_69939247448224
// MI455X (gfx1250) — matrix operand order repaired and checked by reading, not compiled, not run on hardware
//
#include <hip/hip_runtime.h>
#include <hip/hip_bf16.h>
#include <math.h>

// ---------------------------------------------------------------------------
// HGT forward for gfx1250 (MI455X). All dense math on v_wmma_f32_16x16x32_bf16
// (f32 accumulate). Relation transforms are folded into the K/V projection
// weights (one GEMM instead of GEMM+transform); GELU is fused into the Wa GEMM
// A-load and the gated skip + ELU into its epilogue. Weights staged in LDS as
// bf16 transposed so each B fragment is two ds_load_b128.
// ---------------------------------------------------------------------------

#define HDW   128   // H*D hidden width
#define NHEAD 4
#define DHEAD 32
#define LDSK  136   // padded k-stride (bf16 elems) to spread LDS banks

typedef __attribute__((ext_vector_type(16))) __bf16 bfx16;
typedef __attribute__((ext_vector_type(8)))  float  fx8;

union FragAB { bfx16 v; unsigned short u[16]; uint4 q[2]; };
union BPack  { __bf16 b[2]; unsigned u; };

__device__ __forceinline__ float gelu_f(float v) {
  return 0.5f * v * (1.0f + erff(v * 0.7071067811865476f));   // exact GELU
}

// float atomic max via signed/unsigned int ordering trick (init to -FLT_MAX)
__device__ __forceinline__ void atomicMaxFloat(float* addr, float value) {
  if (value >= 0.0f) {
    atomicMax((int*)addr, __float_as_int(value));
  } else {
    atomicMin((unsigned int*)addr, __float_as_uint(value));
  }
}

template <bool GELUA>
__device__ __forceinline__ void packA16(bfx16& v, float4 a0, float4 a1,
                                        float4 a2, float4 a3) {
  float t[16] = {a0.x, a0.y, a0.z, a0.w, a1.x, a1.y, a1.z, a1.w,
                 a2.x, a2.y, a2.z, a2.w, a3.x, a3.y, a3.z, a3.w};
#pragma unroll
  for (int j = 0; j < 16; ++j) {
    float x = t[j];
    if (GELUA) x = gelu_f(x);
    v[j] = (__bf16)x;          // native v_cvt bf16 path
  }
}

// ---------------------------------------------------------------------------
// C[M,128] = A[M,K] @ W[K,128] + bias[128]        (optionally gelu on A)
// If skipv != nullptr:  C = elu(beta*(A@W+b) + (1-beta)*C), beta=sigmoid(*skipv)
// block = 256 (8 waves); each wave owns a 16-row strip and all 8 col tiles.
// W staged per block into LDS as bf16, transposed [n][k], zero-padded to
// roundup32(K).  VEC => A rows are 16B-aligned (K % 4 == 0).
// ---------------------------------------------------------------------------
template <bool VEC, bool GELUA>
__global__ __launch_bounds__(256)
void gemm_bias_kernel(const float* __restrict__ A, const float* __restrict__ W,
                      const float* __restrict__ bias, float* __restrict__ C,
                      int M, int K, const float* __restrict__ skipv) {
  __shared__ unsigned short Wlds[HDW * LDSK];   // 34.8 KB

  const int t    = threadIdx.x;
  const int kpad = (K + 31) & ~31;

  // ---- cooperative stage: Wlds[n][k] = bf16(W[k][n]), zero beyond K -------
  {
    const int n   = t & 127;
    const int kb0 = (t >> 7) * 64;
    for (int j = 0; j < 64; j += 2) {
      int k = kb0 + j;
      if (k >= kpad) break;
      BPack pk;
      pk.b[0] = (__bf16)((k     < K) ? W[(size_t)k * HDW + n]       : 0.0f);
      pk.b[1] = (__bf16)((k + 1 < K) ? W[(size_t)(k + 1) * HDW + n] : 0.0f);
      *(unsigned*)(&Wlds[n * LDSK + k]) = pk.u;
    }
  }
  __syncthreads();

  const int lane = t & 31;
  const int wave = t >> 5;
  const int m    = lane & 15;
  const int half = lane >> 4;                   // 0: lanes 0-15, 1: 16-31
  const int row0 = blockIdx.x * 128 + wave * 16;
  if (row0 >= M) return;

  // clamp instead of guard: out-of-range rows compute garbage never stored
  const int arow  = min(row0 + m, M - 1);
  const float* ap = A + (size_t)arow * K;

  fx8 acc[8];
#pragma unroll
  for (int c = 0; c < 8; ++c) acc[c] = fx8{0.f,0.f,0.f,0.f,0.f,0.f,0.f,0.f};

  const int kfull = K & ~31;
  for (int kb = 0; kb < kfull; kb += 32) {
    FragAB af;
    const int ak0 = kb + half * 8;
    if (VEC) {
      packA16<GELUA>(af.v,
                     *(const float4*)(ap + ak0),      *(const float4*)(ap + ak0 + 4),
                     *(const float4*)(ap + ak0 + 16), *(const float4*)(ap + ak0 + 20));
    } else {
#pragma unroll
      for (int j = 0; j < 8; ++j) {
        float x0 = ap[ak0 + j], x1 = ap[ak0 + 16 + j];
        if (GELUA) { x0 = gelu_f(x0); x1 = gelu_f(x1); }
        af.v[j]     = (__bf16)x0;
        af.v[j + 8] = (__bf16)x1;
      }
    }
#pragma unroll
    for (int c = 0; c < 8; ++c) {
      FragAB bfm;
      const uint4* bp = (const uint4*)(&Wlds[(c * 16 + m) * LDSK + kb + half * 8]);
      bfm.q[0] = bp[0];
      bfm.q[1] = bp[2];
      acc[c] = __builtin_amdgcn_wmma_f32_16x16x32_bf16(false, af.v, false, bfm.v,
                                                       (short)0, acc[c], false, false);
    }
  }

  if (kfull < K) {                              // guarded K tail (input proj only)
    const int kb  = kfull;
    const int ak0 = kb + half * 8;
    const int ak1 = kb + 16 + half * 8;
    FragAB af;
#pragma unroll
    for (int j = 0; j < 8; ++j) {
      float x0 = ((ak0 + j) < K) ? ap[ak0 + j] : 0.0f;
      float x1 = ((ak1 + j) < K) ? ap[ak1 + j] : 0.0f;
      if (GELUA) { x0 = gelu_f(x0); x1 = gelu_f(x1); }
      af.v[j]     = (__bf16)x0;
      af.v[j + 8] = (__bf16)x1;
    }
#pragma unroll
    for (int c = 0; c < 8; ++c) {
      FragAB bfm;
      const uint4* bp = (const uint4*)(&Wlds[(c * 16 + m) * LDSK + kb + half * 8]);
      bfm.q[0] = bp[0];
      bfm.q[1] = bp[2];
      acc[c] = __builtin_amdgcn_wmma_f32_16x16x32_bf16(false, af.v, false, bfm.v,
                                                       (short)0, acc[c], false, false);
    }
  }

  // ---- epilogue: C vgpr i -> row rbase+i, col c*16+m ---------------------
  float beta = 0.0f, omb = 0.0f;
  const bool gated = (skipv != nullptr);
  if (gated) { beta = 1.0f / (1.0f + expf(-*skipv)); omb = 1.0f - beta; }

  const int rbase = row0 + half * 8;
  if (row0 + 16 <= M) {                         // fast path, no guards
#pragma unroll
    for (int c = 0; c < 8; ++c) {
      const int n = c * 16 + m;
      const float bv = bias[n];
#pragma unroll
      for (int i = 0; i < 8; ++i) {
        size_t idx = (size_t)(rbase + i) * HDW + n;
        float r = acc[c][i] + bv;
        if (gated) {
          float v = beta * r + omb * C[idx];
          C[idx] = v > 0.0f ? v : (expf(v) - 1.0f);   // ELU
        } else {
          C[idx] = r;
        }
      }
    }
  } else {
#pragma unroll
    for (int c = 0; c < 8; ++c) {
      const int n = c * 16 + m;
      const float bv = bias[n];
#pragma unroll
      for (int i = 0; i < 8; ++i) {
        int rr = rbase + i;
        if (rr >= M) continue;
        size_t idx = (size_t)rr * HDW + n;
        float r = acc[c][i] + bv;
        if (gated) {
          float v = beta * r + omb * C[idx];
          C[idx] = v > 0.0f ? v : (expf(v) - 1.0f);
        } else {
          C[idx] = r;
        }
      }
    }
  }
}

// ---------------------------------------------------------------------------
// Fold block-diagonal relation matrix into a projection:
//   Wf[k][n] = sum_d W[k][h*32+d] * rel[h][d][n&31],   n = h*32 + f
//   bf[n]    = sum_d b[h*32+d]    * rel[h][d][n&31]
// grid = 128 (rows k), block = 128 (cols n)
// ---------------------------------------------------------------------------
__global__ void fuse_weight_kernel(const float* __restrict__ W,
                                   const float* __restrict__ rel,
                                   const float* __restrict__ bias,
                                   float* __restrict__ Wf,
                                   float* __restrict__ bf) {
  const int k = blockIdx.x;
  const int n = threadIdx.x;
  const int h = n >> 5, f = n & 31;
  const float* wp = W + (size_t)k * HDW + h * DHEAD;
  const float* rp = rel + h * DHEAD * DHEAD + f;
  float acc = 0.0f;
#pragma unroll
  for (int d = 0; d < DHEAD; ++d) acc += wp[d] * rp[(size_t)d * DHEAD];
  Wf[(size_t)k * HDW + n] = acc;
  if (k == 0) {
    const float* bp = bias + h * DHEAD;
    float bacc = 0.0f;
#pragma unroll
    for (int d = 0; d < DHEAD; ++d) bacc += bp[d] * rp[(size_t)d * DHEAD];
    bf[n] = bacc;
  }
}

// ---------------------------------------------------------------------------
__global__ void fill_kernel(float* __restrict__ p, float v, size_t n) {
  size_t i = (size_t)blockIdx.x * blockDim.x + threadIdx.x;
  if (i < n) p[i] = v;
}

// score[e,h] = SCALE * p[h] * dot(q[dst,h,:], kt[src,h,:]);  atomicMax -> mx
__global__ void edge_score_kernel(const float* __restrict__ q,
                                  const float* __restrict__ kt,
                                  const int* __restrict__ src,
                                  const int* __restrict__ dst,
                                  const float* __restrict__ p,
                                  float* __restrict__ score,
                                  float* __restrict__ mx, int E) {
  int t = blockIdx.x * blockDim.x + threadIdx.x;
  if (t >= E * NHEAD) return;
  int e = t >> 2, h = t & 3;
  int s = src[e], d = dst[e];
  const float* qp = q  + (size_t)d * HDW + h * DHEAD;
  const float* kp = kt + (size_t)s * HDW + h * DHEAD;
  float acc = 0.0f;
#pragma unroll
  for (int i = 0; i < DHEAD; i += 4) {
    float4 qa = *(const float4*)(qp + i);
    float4 ka = *(const float4*)(kp + i);
    acc += qa.x * ka.x + qa.y * ka.y + qa.z * ka.z + qa.w * ka.w;
  }
  float sc = acc * p[h] * 0.17677669529663687f;   // 1/sqrt(32)
  score[t] = sc;
  atomicMaxFloat(&mx[(size_t)d * NHEAD + h], sc);
}

// ex = exp(score - mx[dst]);  score <- ex;  atomicAdd -> den
__global__ void edge_exp_kernel(float* __restrict__ score,
                                const float* __restrict__ mx,
                                float* __restrict__ den,
                                const int* __restrict__ dst, int E) {
  int t = blockIdx.x * blockDim.x + threadIdx.x;
  if (t >= E * NHEAD) return;
  int e = t >> 2, h = t & 3;
  int d = dst[e];
  float ex = expf(score[t] - mx[(size_t)d * NHEAD + h]);
  score[t] = ex;
  atomicAdd(&den[(size_t)d * NHEAD + h], ex);
}

// agg[dst, f] += (ex/den[dst,h]) * vt[src, f]; one block per edge, 128 thr.
__global__ void edge_aggregate_kernel(const float* __restrict__ vt,
                                      const float* __restrict__ ex,
                                      const float* __restrict__ den,
                                      const int* __restrict__ src,
                                      const int* __restrict__ dst,
                                      float* __restrict__ agg, int E) {
  int e = blockIdx.x;
  if (e >= E) return;
  int t = threadIdx.x;               // feature 0..127
  int h = t >> 5;
  int s = src[e], d = dst[e];
  float alpha = ex[(size_t)e * NHEAD + h] / den[(size_t)d * NHEAD + h];
  atomicAdd(&agg[(size_t)d * HDW + t], alpha * vt[(size_t)s * HDW + t]);
}

// ---------------------------------------------------------------------------
extern "C" void kernel_launch(void* const* d_in, const int* in_sizes, int n_in,
                              void* d_out, int out_size, void* d_ws,
                              size_t ws_size, hipStream_t stream) {
  (void)n_in; (void)out_size; (void)ws_size;
  const float* x_m     = (const float*)d_in[0];
  const float* x_g     = (const float*)d_in[1];
  const float* W_in_m  = (const float*)d_in[2];
  const float* b_in_m  = (const float*)d_in[3];
  const float* W_in_g  = (const float*)d_in[4];
  const float* b_in_g  = (const float*)d_in[5];
  const float* Wk_m    = (const float*)d_in[6];
  const float* bk_m    = (const float*)d_in[7];
  const float* Wq_m    = (const float*)d_in[8];
  const float* bq_m    = (const float*)d_in[9];
  const float* Wv_m    = (const float*)d_in[10];
  const float* bv_m    = (const float*)d_in[11];
  const float* Wk_g    = (const float*)d_in[12];
  const float* bk_g    = (const float*)d_in[13];
  const float* Wq_g    = (const float*)d_in[14];
  const float* bq_g    = (const float*)d_in[15];
  const float* Wv_g    = (const float*)d_in[16];
  const float* bv_g    = (const float*)d_in[17];
  const float* a_mg    = (const float*)d_in[18];
  const float* m_mg    = (const float*)d_in[19];
  const float* p_mg    = (const float*)d_in[20];
  const float* a_gm    = (const float*)d_in[21];
  const float* m_gm    = (const float*)d_in[22];
  const float* p_gm    = (const float*)d_in[23];
  const float* Wa_m    = (const float*)d_in[24];
  const float* ba_m    = (const float*)d_in[25];
  const float* Wa_g    = (const float*)d_in[26];
  const float* ba_g    = (const float*)d_in[27];
  const float* skip_m  = (const float*)d_in[28];
  const float* skip_g  = (const float*)d_in[29];
  const float* W_out_m = (const float*)d_in[30];
  const float* b_out_m = (const float*)d_in[31];
  const float* W_out_g = (const float*)d_in[32];
  const float* b_out_g = (const float*)d_in[33];
  const int* e_mg_src  = (const int*)d_in[34];
  const int* e_mg_dst  = (const int*)d_in[35];
  const int* e_gm_src  = (const int*)d_in[36];
  const int* e_gm_dst  = (const int*)d_in[37];

  const int Fm  = in_sizes[2] / HDW;
  const int Fg  = in_sizes[4] / HDW;
  const int Nm  = in_sizes[0] / Fm;
  const int Ng  = in_sizes[1] / Fg;
  const int Emg = in_sizes[34];
  const int Egm = in_sizes[36];

  // ---- workspace layout (256B aligned slices) --------------------------
  float* ws = (float*)d_ws;
  size_t off = 0;
  auto carve = [&](size_t n) {
    float* p = ws + off;
    off += (n + 63) & ~(size_t)63;
    return p;
  };
  float* h_m   = carve((size_t)Nm * HDW);
  float* q_m   = carve((size_t)Nm * HDW);
  float* kt_m  = carve((size_t)Nm * HDW);
  float* vt_m  = carve((size_t)Nm * HDW);
  float* agg_m = carve((size_t)Nm * HDW);
  float* h_g   = carve((size_t)Ng * HDW);
  float* q_g   = carve((size_t)Ng * HDW);
  float* kt_g  = carve((size_t)Ng * HDW);
  float* vt_g  = carve((size_t)Ng * HDW);
  float* agg_g = carve((size_t)Ng * HDW);
  float* sc_mg = carve((size_t)Emg * NHEAD);
  float* sc_gm = carve((size_t)Egm * NHEAD);
  float* mx_g  = carve((size_t)Ng * NHEAD);
  float* den_g = carve((size_t)Ng * NHEAD);
  float* mx_m  = carve((size_t)Nm * NHEAD);
  float* den_m = carve((size_t)Nm * NHEAD);
  float* Wf_km = carve((size_t)HDW * HDW);   // fused Wk@blockdiag(a), per layer
  float* Wf_vm = carve((size_t)HDW * HDW);
  float* Wf_kg = carve((size_t)HDW * HDW);
  float* Wf_vg = carve((size_t)HDW * HDW);
  float* bf_km = carve(HDW);
  float* bf_vm = carve(HDW);
  float* bf_kg = carve(HDW);
  float* bf_vg = carve(HDW);

  // ---- launch helpers ---------------------------------------------------
  auto gemm = [&](const float* A, const float* W, const float* b, float* C,
                  int M, int K, bool gelu_a = false,
                  const float* skipv = nullptr) {
    dim3 grid((unsigned)((M + 127) / 128));
    if (gelu_a)
      gemm_bias_kernel<true, true ><<<grid, dim3(256), 0, stream>>>(A, W, b, C, M, K, skipv);
    else if ((K & 3) == 0)
      gemm_bias_kernel<true, false><<<grid, dim3(256), 0, stream>>>(A, W, b, C, M, K, skipv);
    else
      gemm_bias_kernel<false, false><<<grid, dim3(256), 0, stream>>>(A, W, b, C, M, K, skipv);
  };
  auto fuse = [&](const float* W, const float* rel, const float* b,
                  float* Wf, float* bf) {
    fuse_weight_kernel<<<dim3(128), dim3(128), 0, stream>>>(W, rel, b, Wf, bf);
  };
  auto fill = [&](float* p, float v, size_t n) {
    fill_kernel<<<dim3((unsigned)((n + 255) / 256)), dim3(256), 0, stream>>>(p, v, n);
  };
  auto ewise = [&](size_t n) { return dim3((unsigned)((n + 255) / 256)); };

  auto side = [&](const float* q, const float* kt, const float* vt,
                  const int* src, const int* dst, const float* p,
                  float* score, float* mx, float* den, float* agg,
                  int Ndst, int E) {
    fill(agg, 0.0f, (size_t)Ndst * HDW);
    fill(den, 0.0f, (size_t)Ndst * NHEAD);
    fill(mx, -3.402823466e38f, (size_t)Ndst * NHEAD);
    size_t tE = (size_t)E * NHEAD;
    edge_score_kernel<<<ewise(tE), 256, 0, stream>>>(q, kt, src, dst, p,
                                                     score, mx, E);
    edge_exp_kernel<<<ewise(tE), 256, 0, stream>>>(score, mx, den, dst, E);
    edge_aggregate_kernel<<<dim3((unsigned)E), 128, 0, stream>>>(vt, score,
                                                                 den, src,
                                                                 dst, agg, E);
  };

  // ---- input projections ------------------------------------------------
  gemm(x_m, W_in_m, b_in_m, h_m, Nm, Fm);
  gemm(x_g, W_in_g, b_in_g, h_g, Ng, Fg);

  // ---- HGT layers -------------------------------------------------------
  for (int l = 0; l < 2; ++l) {
    const size_t wo = (size_t)l * HDW * HDW;
    const size_t bo = (size_t)l * HDW;
    const size_t ro = (size_t)l * NHEAD * DHEAD * DHEAD;
    const size_t po = (size_t)l * NHEAD;

    // fold relation matrices into K/V projection weights (tiny kernels)
    fuse(Wk_m + wo, a_mg + ro, bk_m + bo, Wf_km, bf_km);
    fuse(Wv_m + wo, m_mg + ro, bv_m + bo, Wf_vm, bf_vm);
    fuse(Wk_g + wo, a_gm + ro, bk_g + bo, Wf_kg, bf_kg);
    fuse(Wv_g + wo, m_gm + ro, bv_g + bo, Wf_vg, bf_vg);

    gemm(h_m, Wq_m + wo, bq_m + bo, q_m, Nm, HDW);
    gemm(h_m, Wf_km, bf_km, kt_m, Nm, HDW);     // kt directly
    gemm(h_m, Wf_vm, bf_vm, vt_m, Nm, HDW);
    gemm(h_g, Wq_g + wo, bq_g + bo, q_g, Ng, HDW);
    gemm(h_g, Wf_kg, bf_kg, kt_g, Ng, HDW);
    gemm(h_g, Wf_vg, bf_vg, vt_g, Ng, HDW);

    // attention + aggregation
    side(q_g, kt_m, vt_m, e_mg_src, e_mg_dst, p_mg + po,
         sc_mg, mx_g, den_g, agg_g, Ng, Emg);
    side(q_m, kt_g, vt_g, e_gm_src, e_gm_dst, p_gm + po,
         sc_gm, mx_m, den_m, agg_m, Nm, Egm);

    // h = elu(beta*(gelu(agg)@Wa+ba) + (1-beta)*h)  -- fully fused GEMM
    gemm(agg_m, Wa_m + wo, ba_m + bo, h_m, Nm, HDW, true, skip_m + l);
    gemm(agg_g, Wa_g + wo, ba_g + bo, h_g, Ng, HDW, true, skip_g + l);
  }

  // ---- output projections ----------------------------------------------
  float* out = (float*)d_out;
  gemm(h_m, W_out_m, b_out_m, out, Nm, HDW);
  gemm(h_g, W_out_g, b_out_g, out + (size_t)Nm * HDW, Ng, HDW);
}
